// DirectGPT2Model_81913616269371
// MI455X (gfx1250) — compile-verified, matrix operand order checked by reading, not run on hardware
//
#include <hip/hip_runtime.h>
#include <hip/hip_bf16.h>
#include <math.h>

typedef __attribute__((ext_vector_type(16))) _Float16 v16h;
typedef __attribute__((ext_vector_type(8)))  float    v8f;
typedef __attribute__((ext_vector_type(4)))  unsigned int v4u;
typedef __attribute__((ext_vector_type(4)))  int v4i;

#define NTOK 4096      // B*S = 2*2048
#define SEQ  2048
#define EDIM 1024
#define NHEAD 16
#define HD   64

// gfx1250 async global->LDS copy path (ASYNCcnt-tracked), with sync fallback
#if defined(__gfx1250__) && \
    __has_builtin(__builtin_amdgcn_global_load_async_to_lds_b128) && \
    __has_builtin(__builtin_amdgcn_s_wait_asynccnt)
#define USE_ASYNC_LDS 1
#else
#define USE_ASYNC_LDS 0
#endif

// ---------------------------------------------------------------------------
// f32 -> f16 conversion (weights)
// ---------------------------------------------------------------------------
__global__ void __launch_bounds__(256) cvt_f32_f16(const float* __restrict__ in,
                                                   _Float16* __restrict__ out, int n) {
    int g = blockIdx.x * 256 + threadIdx.x;
    if (g < n) out[g] = (_Float16)in[g];
}

// ---------------------------------------------------------------------------
// LayerNorm (E=1024) + cast to f16. One block (256 thr) per row.
// ---------------------------------------------------------------------------
__global__ void __launch_bounds__(256) ln_cast_kernel(const float* __restrict__ x,
                                                      const float* __restrict__ w,
                                                      const float* __restrict__ b,
                                                      _Float16* __restrict__ out) {
    __shared__ float red[256];
    const int row = blockIdx.x, tid = threadIdx.x;
    const float* xr = x + (size_t)row * EDIM;
    float v[4]; float s = 0.f;
#pragma unroll
    for (int i = 0; i < 4; ++i) { v[i] = xr[tid + 256 * i]; s += v[i]; }
    red[tid] = s; __syncthreads();
    for (int o = 128; o > 0; o >>= 1) { if (tid < o) red[tid] += red[tid + o]; __syncthreads(); }
    const float m = red[0] * (1.f / EDIM);
    __syncthreads();
    s = 0.f;
#pragma unroll
    for (int i = 0; i < 4; ++i) { float d = v[i] - m; s += d * d; }
    red[tid] = s; __syncthreads();
    for (int o = 128; o > 0; o >>= 1) { if (tid < o) red[tid] += red[tid + o]; __syncthreads(); }
    const float rstd = rsqrtf(red[0] * (1.f / EDIM) + 1e-5f);
#pragma unroll
    for (int i = 0; i < 4; ++i) {
        int idx = tid + 256 * i;
        out[(size_t)row * EDIM + idx] = (_Float16)((v[i] - m) * rstd * w[idx] + b[idx]);
    }
}

// ---------------------------------------------------------------------------
// WMMA GEMM: C(MxN) = A(MxK,f16) * B(KxN,f16) + bias [+ residual]
// Block tile 128x128, K-step 32, 8 waves (256 thr), each wave 32x64 = 8 WMMAs.
// Double-buffered LDS; A tile streamed via async global->LDS (gfx1250 path),
// B tile staged transposed as 32-bit (k,k+1) pairs.
// ---------------------------------------------------------------------------
#define BM 128
#define BN 128
#define BK 32
#define LPAD 8

union HFrag { v16h v; _Float16 h[16]; };
union FAcc  { v8f  v; float    f[8];  };
union HPair { _Float16 h2[2]; unsigned u; };

__device__ __forceinline__ void stage_tile(const _Float16* __restrict__ A,
                                           const _Float16* __restrict__ B,
                                           _Float16 (*As)[BK + LPAD],
                                           _Float16 (*Bs)[BK + LPAD],
                                           int blockRow, int blockCol,
                                           int K, int Nc, int kt, int tid) {
    // ---- B tile: load two consecutive K rows, store interleaved 32-bit pairs
    {
        int pr = tid >> 4;                 // pair-row 0..15  (k = 2*pr, 2*pr+1)
        int n8 = (tid & 15) << 3;          // 8 consecutive columns
        const _Float16* b0 = B + (size_t)(kt * BK + 2 * pr) * Nc + blockCol + n8;
        v4u lo = *(const v4u*)b0;
        v4u hi = *(const v4u*)(b0 + Nc);
        const _Float16* lv = (const _Float16*)&lo;
        const _Float16* hv = (const _Float16*)&hi;
#pragma unroll
        for (int j = 0; j < 8; ++j) {
            HPair p; p.h2[0] = lv[j]; p.h2[1] = hv[j];
            *(unsigned*)&Bs[n8 + j][2 * pr] = p.u;
        }
    }
    // ---- A tile: bulk layout-preserving copy, 2 x 16B per thread ----
#pragma unroll
    for (int i = 0; i < 2; ++i) {
        int idx = tid * 2 + i;             // 0..511
        int r   = idx >> 2;
        int k8  = (idx & 3) << 3;
        const _Float16* gp = A + (size_t)(blockRow + r) * K + kt * BK + k8;
        _Float16* lp = &As[r][k8];
#if USE_ASYNC_LDS
        __builtin_amdgcn_global_load_async_to_lds_b128(
            (__attribute__((address_space(1))) v4i*)gp,
            (__attribute__((address_space(3))) v4i*)lp, 0, 0);
#else
        *(v4u*)lp = *(const v4u*)gp;
#endif
    }
}

__global__ void __launch_bounds__(256) wmma_gemm_kernel(const _Float16* __restrict__ A,
                                                        const _Float16* __restrict__ B,
                                                        const float* __restrict__ bias,
                                                        const float* residual,
                                                        float* C, int M, int Nc, int K) {
    __shared__ _Float16 As[2][BM][BK + LPAD];   // [buf][m][k]
    __shared__ _Float16 Bs[2][BN][BK + LPAD];   // [buf][n][k] (transposed)

    const int tid   = threadIdx.x;
    const int lane  = tid & 31;
    const int wave  = tid >> 5;
    const int waveM = wave >> 1;                // 0..3 -> rows
    const int waveN = wave & 1;                 // 0..1 -> cols
    const int rsel  = lane & 15;
    const int hsel  = lane >> 4;

    const int blockRow = blockIdx.y * BM;
    const int blockCol = blockIdx.x * BN;

    FAcc acc[2][4];
#pragma unroll
    for (int mi = 0; mi < 2; ++mi)
#pragma unroll
        for (int ni = 0; ni < 4; ++ni) acc[mi][ni].v = (v8f){};

    const int tiles = K / BK;

    // prologue: stage tile 0 into buffer 0
    stage_tile(A, B, As[0], Bs[0], blockRow, blockCol, K, Nc, 0, tid);
#if USE_ASYNC_LDS
    __builtin_amdgcn_s_wait_asynccnt(0);
#endif
    __syncthreads();

    for (int kt = 0; kt < tiles; ++kt) {
        const int cur = kt & 1;
        // stream next tile into the other buffer while computing this one
        if (kt + 1 < tiles) {
            stage_tile(A, B, As[cur ^ 1], Bs[cur ^ 1], blockRow, blockCol, K, Nc, kt + 1, tid);
            __builtin_prefetch(B + (size_t)((kt + 1) * BK + (tid >> 4)) * Nc + blockCol, 0, 1);
        }

        // ---- build fragments per ISA wave32 16-bit layouts ----
        HFrag af[2], bf[4];
#pragma unroll
        for (int mi = 0; mi < 2; ++mi) {
            const _Float16* ar = &As[cur][waveM * 32 + mi * 16 + rsel][0];
#pragma unroll
            for (int v = 0; v < 8; ++v) {
                int k = ((v < 4) ? (v * 2) : (16 + (v - 4) * 2)) + hsel * 8;
                af[mi].h[2 * v]     = ar[k];
                af[mi].h[2 * v + 1] = ar[k + 1];
            }
        }
#pragma unroll
        for (int ni = 0; ni < 4; ++ni) {
            const _Float16* br = &Bs[cur][waveN * 64 + ni * 16 + rsel][0];
#pragma unroll
            for (int v = 0; v < 8; ++v) {
                int k = ((v < 4) ? (v * 2) : (16 + (v - 4) * 2)) + hsel * 8;
                bf[ni].h[2 * v]     = br[k];
                bf[ni].h[2 * v + 1] = br[k + 1];
            }
        }
#pragma unroll
        for (int mi = 0; mi < 2; ++mi)
#pragma unroll
            for (int ni = 0; ni < 4; ++ni)
                acc[mi][ni].v = __builtin_amdgcn_wmma_f32_16x16x32_f16(
                    false, af[mi].v, false, bf[ni].v, (short)0, acc[mi][ni].v, false, false);

#if USE_ASYNC_LDS
        __builtin_amdgcn_s_wait_asynccnt(0);   // next-buffer async writes landed
#endif
        __syncthreads();
    }

    // ---- epilogue: C = acc + bias (+ residual) ----
#pragma unroll
    for (int mi = 0; mi < 2; ++mi) {
#pragma unroll
        for (int ni = 0; ni < 4; ++ni) {
            int col = blockCol + waveN * 64 + ni * 16 + rsel;
            int rb  = blockRow + waveM * 32 + mi * 16 + 8 * hsel;
#pragma unroll
            for (int vg = 0; vg < 8; ++vg) {
                int row = rb + vg;
                float val = acc[mi][ni].f[vg] + bias[col];
                size_t o = (size_t)row * Nc + col;
                if (residual) val += residual[o];
                C[o] = val;
            }
        }
    }
}

// ---------------------------------------------------------------------------
// Quaternion build + double Hamilton product, component 0 only.
// qkv layout per token: [h*192 + {q:0,k:64,v:128} + d]
// ---------------------------------------------------------------------------
__global__ void __launch_bounds__(256) quat_kernel(const float* __restrict__ qkv,
                                                   float* __restrict__ p0) {
    int g   = blockIdx.x * 256 + threadIdx.x;     // 0 .. NTOK*1024
    int tok = g >> 10;
    int e   = g & 1023;
    int h   = e >> 6, d = e & 63;
    int s   = tok & (SEQ - 1);
    int hm1 = (h + NHEAD - 1) & (NHEAD - 1);
    size_t base  = (size_t)tok * 3072;
    size_t baseP = (size_t)(tok - 1) * 3072;

    auto load4 = [&](int off, float* c) {
        float t = qkv[base + h * 192 + off + d];
        c[0] = t;
        c[1] = (s > 0) ? (t - qkv[baseP + h * 192 + off + d]) : 0.f;
        c[2] = qkv[base + hm1 * 192 + off + d];
        c[3] = cosf(t);
    };
    float q[4], k[4], v[4];
    load4(0, q); load4(64, k); load4(128, v);

    float qk0 = q[0]*k[0] - q[1]*k[1] - q[2]*k[2] - q[3]*k[3];
    float qk1 = q[0]*k[1] + q[1]*k[0] + q[2]*k[3] - q[3]*k[2];
    float qk2 = q[0]*k[2] - q[1]*k[3] + q[2]*k[0] + q[3]*k[1];
    float qk3 = q[0]*k[3] + q[1]*k[2] - q[2]*k[1] + q[3]*k[0];

    p0[(size_t)tok * EDIM + e] = qk0*v[0] - qk1*v[1] - qk2*v[2] - qk3*v[3];
}

// ---------------------------------------------------------------------------
// Spectral kernel: per-token 16x64 DFT -> filter -> iDFT -> real, modulation,
// cast to f16. One block (256 thr) per token; 4 outputs per thread per stage.
// ---------------------------------------------------------------------------
__global__ void __launch_bounds__(256) spectral_kernel(const float* __restrict__ p0,
                                                       _Float16* __restrict__ out) {
    __shared__ float xr[NHEAD][HD];
    __shared__ float fr[NHEAD][HD], fi[NHEAD][HD];
    __shared__ float gr[NHEAD][HD], gi[NHEAD][HD];
    const int tok = blockIdx.x;
    const int s   = tok & (SEQ - 1);
    const int tid = threadIdx.x;
    const float TWO_PI = 6.28318530717958647692f;
    const float* src = p0 + (size_t)tok * EDIM;

#pragma unroll
    for (int j = 0; j < 4; ++j) { int f = tid + 256 * j; xr[f >> 6][f & 63] = src[f]; }
    __syncthreads();

    // forward DFT along d (64)
#pragma unroll
    for (int j = 0; j < 4; ++j) {
        int f = tid + 256 * j; int h = f >> 6, kd = f & 63;
        float sn, cs; __sincosf(-TWO_PI * (float)kd * (1.f / 64.f), &sn, &cs);
        float cr = 1.f, ci = 0.f, ar = 0.f, ai = 0.f;
        for (int dd = 0; dd < 64; ++dd) {
            float xv = xr[h][dd];
            ar += xv * cr; ai += xv * ci;
            float nr = cr * cs - ci * sn; ci = cr * sn + ci * cs; cr = nr;
        }
        fr[h][kd] = ar; fi[h][kd] = ai;
    }
    __syncthreads();

    // forward DFT along h (16) + spectral filter
#pragma unroll
    for (int j = 0; j < 4; ++j) {
        int f = tid + 256 * j; int kh = f >> 6, kd = f & 63;
        float sn, cs; __sincosf(-TWO_PI * (float)kh * (1.f / 16.f), &sn, &cs);
        float cr = 1.f, ci = 0.f, ar = 0.f, ai = 0.f;
        for (int hh = 0; hh < 16; ++hh) {
            float br_ = fr[hh][kd], bi_ = fi[hh][kd];
            ar += br_ * cr - bi_ * ci; ai += br_ * ci + bi_ * cr;
            float nr = cr * cs - ci * sn; ci = cr * sn + ci * cs; cr = nr;
        }
        float ks  = ((s  < 1024) ? (float)s  : (float)(s  - 2048)) * (1.f / 2048.f);
        float khf = ((kh < 8)    ? (float)kh : (float)(kh - 16))   * (1.f / 16.f);
        float kdf = ((kd < 32)   ? (float)kd : (float)(kd - 64))   * (1.f / 64.f);
        float km  = sqrtf(ks * ks + khf * khf + kdf * kdf + 1e-10f);
        float th  = atanf(logf(km + 1e-10f));
        float fsi, fco; __sincosf(th, &fsi, &fco);
        gr[kh][kd] = ar * fco - ai * fsi;
        gi[kh][kd] = ar * fsi + ai * fco;
    }
    __syncthreads();

    // inverse DFT along h
#pragma unroll
    for (int j = 0; j < 4; ++j) {
        int f = tid + 256 * j; int h = f >> 6, kd = f & 63;
        float sn, cs; __sincosf(TWO_PI * (float)h * (1.f / 16.f), &sn, &cs);
        float cr = 1.f, ci = 0.f, ar = 0.f, ai = 0.f;
        for (int kh = 0; kh < 16; ++kh) {
            float br_ = gr[kh][kd], bi_ = gi[kh][kd];
            ar += br_ * cr - bi_ * ci; ai += br_ * ci + bi_ * cr;
            float nr = cr * cs - ci * sn; ci = cr * sn + ci * cs; cr = nr;
        }
        fr[h][kd] = ar; fi[h][kd] = ai;
    }
    __syncthreads();

    // inverse DFT along d, real part, 1/1024 norm, modulation, cast f16
    float lam = (float)s * (1.f / (float)SEQ);
    float t0  = 0.1f * (float)s;
    float md  = sinf(t0 + lam) * cosf(t0 - 2.f * lam + 0.5f * lam * lam);
    float scale = (1.f + 0.1f * md) * (1.f / 1024.f);
#pragma unroll
    for (int j = 0; j < 4; ++j) {
        int f = tid + 256 * j; int h = f >> 6, d = f & 63;
        float sn, cs; __sincosf(TWO_PI * (float)d * (1.f / 64.f), &sn, &cs);
        float cr = 1.f, ci = 0.f, re = 0.f;
        for (int kd = 0; kd < 64; ++kd) {
            re += fr[h][kd] * cr - fi[h][kd] * ci;
            float nr = cr * cs - ci * sn; ci = cr * sn + ci * cs; cr = nr;
        }
        out[(size_t)tok * EDIM + f] = (_Float16)(re * scale);
    }
}

// ---------------------------------------------------------------------------
// exact GELU + cast f16
// ---------------------------------------------------------------------------
__global__ void __launch_bounds__(256) gelu_cast_kernel(const float* __restrict__ in,
                                                        _Float16* __restrict__ out) {
    int g = blockIdx.x * 256 + threadIdx.x;
    float v = in[g];
    out[g] = (_Float16)(0.5f * v * (1.f + erff(v * 0.70710678118654752440f)));
}

// ---------------------------------------------------------------------------
// host-side orchestration
// ---------------------------------------------------------------------------
extern "C" void kernel_launch(void* const* d_in, const int* in_sizes, int n_in,
                              void* d_out, int out_size, void* d_ws, size_t ws_size,
                              hipStream_t stream) {
    (void)in_sizes; (void)n_in; (void)out_size; (void)ws_size;
    const float* x_in    = (const float*)d_in[0];
    const float* ln1_w   = (const float*)d_in[1];
    const float* ln1_b   = (const float*)d_in[2];
    const float* attn_w  = (const float*)d_in[3];
    const float* attn_b  = (const float*)d_in[4];
    const float* aproj_w = (const float*)d_in[5];
    const float* aproj_b = (const float*)d_in[6];
    const float* ln2_w   = (const float*)d_in[7];
    const float* ln2_b   = (const float*)d_in[8];
    const float* fc_w    = (const float*)d_in[9];
    const float* fc_b    = (const float*)d_in[10];
    const float* mproj_w = (const float*)d_in[11];
    const float* mproj_b = (const float*)d_in[12];

    float* x = (float*)d_out;                       // activations live in d_out
    (void)hipMemcpyAsync(x, x_in, (size_t)NTOK * EDIM * sizeof(float),
                         hipMemcpyDeviceToDevice, stream);

    // workspace carve-up (bytes)
    char* ws = (char*)d_ws;
    size_t off = 0;
    auto carve = [&](size_t bytes) { void* p = ws + off; off += (bytes + 255) & ~(size_t)255; return p; };
    _Float16* wqkv_h = (_Float16*)carve((size_t)EDIM * 3 * EDIM * 2);
    _Float16* wprj_h = (_Float16*)carve((size_t)EDIM * EDIM * 2);
    _Float16* wfc_h  = (_Float16*)carve((size_t)EDIM * 4 * EDIM * 2);
    _Float16* wmp_h  = (_Float16*)carve((size_t)4 * EDIM * EDIM * 2);
    _Float16* xh     = (_Float16*)carve((size_t)NTOK * EDIM * 2);
    char*     pool   = (char*)carve((size_t)NTOK * 4 * EDIM * 4);      // qkv+p0 | hmid
    _Float16* ah     = (_Float16*)carve((size_t)NTOK * EDIM * 2);
    _Float16* hh     = (_Float16*)carve((size_t)NTOK * 4 * EDIM * 2);
    float* qkv  = (float*)pool;                                        // NTOK x 3072
    float* p0   = (float*)(pool + (size_t)NTOK * 3 * EDIM * 4);        // NTOK x 1024
    float* hmid = (float*)pool;                                        // NTOK x 4096 (aliases)

    const int nW1 = EDIM * 3 * EDIM, nW2 = EDIM * EDIM, nW3 = EDIM * 4 * EDIM;

    for (int l = 0; l < 4; ++l) {
        // convert this layer's weights to f16
        cvt_f32_f16<<<(nW1 + 255) / 256, 256, 0, stream>>>(attn_w  + (size_t)l * nW1, wqkv_h, nW1);
        cvt_f32_f16<<<(nW2 + 255) / 256, 256, 0, stream>>>(aproj_w + (size_t)l * nW2, wprj_h, nW2);
        cvt_f32_f16<<<(nW3 + 255) / 256, 256, 0, stream>>>(fc_w    + (size_t)l * nW3, wfc_h,  nW3);
        cvt_f32_f16<<<(nW3 + 255) / 256, 256, 0, stream>>>(mproj_w + (size_t)l * nW3, wmp_h,  nW3);

        // attention branch
        ln_cast_kernel<<<NTOK, 256, 0, stream>>>(x, ln1_w + l * EDIM, ln1_b + l * EDIM, xh);
        wmma_gemm_kernel<<<dim3(3 * EDIM / BN, NTOK / BM), 256, 0, stream>>>(
            xh, wqkv_h, attn_b + (size_t)l * 3 * EDIM, nullptr, qkv, NTOK, 3 * EDIM, EDIM);
        quat_kernel<<<NTOK * EDIM / 256, 256, 0, stream>>>(qkv, p0);
        spectral_kernel<<<NTOK, 256, 0, stream>>>(p0, ah);
        wmma_gemm_kernel<<<dim3(EDIM / BN, NTOK / BM), 256, 0, stream>>>(
            ah, wprj_h, aproj_b + (size_t)l * EDIM, x, x, NTOK, EDIM, EDIM);

        // MLP branch
        ln_cast_kernel<<<NTOK, 256, 0, stream>>>(x, ln2_w + l * EDIM, ln2_b + l * EDIM, xh);
        wmma_gemm_kernel<<<dim3(4 * EDIM / BN, NTOK / BM), 256, 0, stream>>>(
            xh, wfc_h, fc_b + (size_t)l * 4 * EDIM, nullptr, hmid, NTOK, 4 * EDIM, EDIM);
        gelu_cast_kernel<<<NTOK * 4 * EDIM / 256, 256, 0, stream>>>(hmid, hh);
        wmma_gemm_kernel<<<dim3(EDIM / BN, NTOK / BM), 256, 0, stream>>>(
            hh, wmp_h, mproj_b + (size_t)l * EDIM, x, x, NTOK, EDIM, 4 * EDIM);
    }
}
